// PNA_10677288698293
// MI455X (gfx1250) — matrix operand order repaired and checked by reading, not compiled, not run on hardware
//
#include <hip/hip_runtime.h>

#define NN 50000
#define EE 800000
#define CC 128
#define TT 4
#define FI 32
#define DEPTH 4
#define HH1 85
#define HH2 42
#define NCLS 10

typedef __attribute__((ext_vector_type(16))) __bf16 v16bf;
typedef __attribute__((ext_vector_type(8)))  __bf16 v8bf;
typedef __attribute__((ext_vector_type(8)))  float  v8f;

static __device__ __forceinline__ unsigned short f2bfbits(float f) {
    unsigned u = __float_as_uint(f);
    unsigned r = u + 0x7FFFu + ((u >> 16) & 1u);
    return (unsigned short)(r >> 16);
}
static __device__ __forceinline__ __bf16 f2bf(float f) {
    unsigned short s = f2bfbits(f);
    union { unsigned short u; __bf16 b; } cv; cv.u = s; return cv.b;
}
// order-preserving float<->uint key for atomic min/max
static __device__ __forceinline__ unsigned fkey(float f) {
    unsigned u = __float_as_uint(f);
    return (u >> 31) ? ~u : (u | 0x80000000u);
}
static __device__ __forceinline__ float funkey(unsigned k) {
    unsigned u = (k >> 31) ? (k & 0x7FFFFFFFu) : ~k;
    return __uint_as_float(u);
}
static __device__ __forceinline__ v16bf cat16(v8bf a, v8bf b) {
    return __builtin_shufflevector(a, b, 0,1,2,3,4,5,6,7,8,9,10,11,12,13,14,15);
}
static __device__ __forceinline__ v8f wmma_bf16(v16bf a, v16bf b, v8f c) {
    return __builtin_amdgcn_wmma_f32_16x16x32_bf16(false, a, false, b, (short)0, c, false, false);
}
static __device__ __forceinline__ v8bf ldv8(const unsigned short* p, unsigned off) {
    return *(const v8bf*)(p + off);
}
// tile index of this wave, forced into an SGPR so tile-level control flow is scalar
static __device__ __forceinline__ int wave_tile() {
    int w = (int)((blockIdx.x * blockDim.x + threadIdx.x) >> 5);
    return __builtin_amdgcn_readfirstlane(w);
}

// ---------------- setup kernels ----------------
__global__ void k_init(float* deg, float* scal, int n) {
    int i = blockIdx.x * blockDim.x + threadIdx.x;
    if (i < n) deg[i] = 0.f;
    if (i == 0) scal[0] = 0.f;
}
__global__ void k_deg(const int* dst, float* deg, int e) {
    int i = blockIdx.x * blockDim.x + threadIdx.x;
    if (i < e) atomicAdd(&deg[dst[i]], 1.0f);
}
__global__ void k_avglog(const float* deg, float* scal, int n) {
    __shared__ float sh[256];
    int i = blockIdx.x * 256 + threadIdx.x;
    sh[threadIdx.x] = (i < n) ? logf(deg[i] + 1.0f) : 0.f;
    __syncthreads();
    for (int s = 128; s > 0; s >>= 1) {
        if (threadIdx.x < s) sh[threadIdx.x] += sh[threadIdx.x + s];
        __syncthreads();
    }
    if (threadIdx.x == 0) atomicAdd(&scal[0], sh[0] * (1.0f / n));
}

// shuffle weights into WMMA B-operand lane order (bf16)
// B operand: lane holds column n = lane&15, K = (lane>=16?8:0)+(j&7)+(j>=8?16:0) within a K=32 step
__global__ void k_prep_pre(const float* W, unsigned short* B) { // [4][4][64][32]
    int i = blockIdx.x * blockDim.x + threadIdx.x;
    if (i >= DEPTH * TT * 2 * 2 * 512) return;
    int j = i & 15, lane = (i >> 4) & 31, nt = (i >> 9) & 1, ks = (i >> 10) & 1;
    int t = (i >> 11) & 3, l = (i >> 13) & 3;
    int n = nt * 16 + (lane & 15);
    int k = ks * 32 + ((lane & 16) ? 8 : 0) + (j & 7) + ((j & 8) ? 16 : 0);
    B[i] = f2bfbits(W[((l * TT + t) * 64 + k) * 32 + n]);
}
__global__ void k_prep_post(const float* W, unsigned short* B) { // [4][4][416][32]
    int i = blockIdx.x * blockDim.x + threadIdx.x;
    if (i >= DEPTH * TT * 13 * 2 * 512) return;
    int j = i & 15, lane = (i >> 4) & 31, nt = (i >> 9) & 1;
    int r = i >> 10; int ks = r % 13; r /= 13; int t = r & 3; int l = r >> 2;
    int n = nt * 16 + (lane & 15);
    int k = ks * 32 + ((lane & 16) ? 8 : 0) + (j & 7) + ((j & 8) ? 16 : 0);
    B[i] = f2bfbits(W[((l * TT + t) * 416 + k) * 32 + n]);
}
__global__ void k_prep_lin(const float* W, unsigned short* B) { // [4][128][128]
    int i = blockIdx.x * blockDim.x + threadIdx.x;
    if (i >= DEPTH * 4 * 8 * 512) return;
    int j = i & 15, lane = (i >> 4) & 31, nt = (i >> 9) & 7, ks = (i >> 12) & 3, l = (i >> 14) & 3;
    int n = nt * 16 + (lane & 15);
    int k = ks * 32 + ((lane & 16) ? 8 : 0) + (j & 7) + ((j & 8) ? 16 : 0);
    B[i] = f2bfbits(W[(l * 128 + k) * 128 + n]);
}

// per-layer: bf16-convert x, reset aggregation surfaces
__global__ void k_cvt_init(const float* x, unsigned short* xb, float* sum, float* sumsq,
                           unsigned* mnk, unsigned* mxk, int total) {
    int i = blockIdx.x * blockDim.x + threadIdx.x;
    if (i >= total) return;
    xb[i] = f2bfbits(x[i]);
    sum[i] = 0.f; sumsq[i] = 0.f;
    mnk[i] = 0xFFFFFFFFu; mxk[i] = 0u;
}

// ---------------- edge message GEMM + multi-aggregate scatter ----------------
__global__ void k_msg(const unsigned short* __restrict__ xb, const unsigned short* __restrict__ Bop,
                      const float* __restrict__ bpre, const int* __restrict__ src,
                      const int* __restrict__ dst, float* __restrict__ sum,
                      float* __restrict__ sumsq, unsigned* __restrict__ mnk,
                      unsigned* __restrict__ mxk, int E, int l) {
    int tile = wave_tile();                 // SGPR
    int lane = threadIdx.x & 31;
    int base = tile * 16;                   // SGPR
    if (base >= E) return;
    int mrow = lane & 15;
    int half = lane >> 4;
    int kb = half * 8;
    bool full = (base + 16 <= E);           // scalar -> s_cbranch
    int eA = min(base + mrow, E - 1);
    unsigned xdoff = (unsigned)dst[eA] * CC;
    unsigned xsoff = (unsigned)src[eA] * CC;
    int nvalid = min(E - base - half * 8, 8);   // rows this half-lane owns
    unsigned rowb[8];
#pragma unroll
    for (int r = 0; r < 8; r++) {
        int e = base + half * 8 + r;
        rowb[r] = (unsigned)dst[min(e, E - 1)] * CC;
    }
#pragma unroll
    for (int t = 0; t < TT; t++) {
        unsigned axd = xdoff + t * 32 + kb;
        unsigned axs = xsoff + t * 32 + kb;
        v16bf aD = cat16(ldv8(xb, axd), ldv8(xb, axd + 16));
        v16bf aS = cat16(ldv8(xb, axs), ldv8(xb, axs + 16));
#pragma unroll
        for (int nt = 0; nt < 2; nt++) {
            unsigned bo = (unsigned)((((l * TT + t) * 2 + 0) * 2 + nt) * 512 + lane * 16);
            v16bf B0 = cat16(ldv8(Bop, bo), ldv8(Bop, bo + 8));
            v16bf B1 = cat16(ldv8(Bop, bo + 1024), ldv8(Bop, bo + 1032));
            v8f c = {};
            c = wmma_bf16(aD, B0, c);   // K = 0..31  -> x_dst
            c = wmma_bf16(aS, B1, c);   // K = 32..63 -> x_src
            int nloc = nt * 16 + mrow;
            float bias = bpre[(l * TT + t) * 32 + nloc];
            unsigned col = (unsigned)(t * 32 + nloc);
            if (full) {                 // scalar fast path, no EXEC churn
#pragma unroll
                for (int r = 0; r < 8; r++) {
                    float v = c[r] + bias;
                    unsigned a = rowb[r] + col;
                    atomicAdd(sum + a, v);
                    atomicAdd(sumsq + a, v * v);
                    unsigned k = fkey(v);
                    atomicMax(mxk + a, k);
                    atomicMin(mnk + a, k);
                }
            } else {
#pragma unroll
                for (int r = 0; r < 8; r++) {
                    if (r >= nvalid) break;
                    float v = c[r] + bias;
                    unsigned a = rowb[r] + col;
                    atomicAdd(sum + a, v);
                    atomicAdd(sumsq + a, v * v);
                    unsigned k = fkey(v);
                    atomicMax(mxk + a, k);
                    atomicMin(mnk + a, k);
                }
            }
        }
    }
}

// ---------------- post-tower MLP (A operand built on the fly) ----------------
static __device__ __forceinline__ void feat8(int sec, unsigned a,
        const float* __restrict__ xcur, const float* __restrict__ sum,
        const float* __restrict__ sumsq, const unsigned* __restrict__ mnk,
        const unsigned* __restrict__ mxk,
        float degv, float inv, float s1, float s2, float* o) {
    if (sec == 0) {
        float4 u = *(const float4*)(xcur + a);
        float4 w = *(const float4*)(xcur + a + 4);
        o[0]=u.x; o[1]=u.y; o[2]=u.z; o[3]=u.w; o[4]=w.x; o[5]=w.y; o[6]=w.z; o[7]=w.w;
        return;
    }
    int ag = (sec - 1) & 3, sc = (sec - 1) >> 2;
    float scale = (sc == 0) ? 1.f : ((sc == 1) ? s1 : s2);
    float vv[8];
    if (ag == 0) { // mean
        float4 u = *(const float4*)(sum + a), w = *(const float4*)(sum + a + 4);
        vv[0]=u.x; vv[1]=u.y; vv[2]=u.z; vv[3]=u.w; vv[4]=w.x; vv[5]=w.y; vv[6]=w.z; vv[7]=w.w;
#pragma unroll
        for (int j = 0; j < 8; j++) vv[j] *= inv;
    } else if (ag == 1) { // min
        uint4 u = *(const uint4*)(mnk + a), w = *(const uint4*)(mnk + a + 4);
        unsigned kk[8] = {u.x,u.y,u.z,u.w,w.x,w.y,w.z,w.w};
#pragma unroll
        for (int j = 0; j < 8; j++) vv[j] = (degv > 0.f) ? funkey(kk[j]) : 0.f;
    } else if (ag == 2) { // max
        uint4 u = *(const uint4*)(mxk + a), w = *(const uint4*)(mxk + a + 4);
        unsigned kk[8] = {u.x,u.y,u.z,u.w,w.x,w.y,w.z,w.w};
#pragma unroll
        for (int j = 0; j < 8; j++) vv[j] = (degv > 0.f) ? funkey(kk[j]) : 0.f;
    } else { // std
        float4 u = *(const float4*)(sum + a), w = *(const float4*)(sum + a + 4);
        float4 q = *(const float4*)(sumsq + a), p = *(const float4*)(sumsq + a + 4);
        float sm[8] = {u.x,u.y,u.z,u.w,w.x,w.y,w.z,w.w};
        float sq[8] = {q.x,q.y,q.z,q.w,p.x,p.y,p.z,p.w};
#pragma unroll
        for (int j = 0; j < 8; j++) {
            float m = sm[j] * inv;
            vv[j] = sqrtf(fmaxf(sq[j] * inv - m * m, 0.f) + 1e-5f);
        }
    }
#pragma unroll
    for (int j = 0; j < 8; j++) o[j] = vv[j] * scale;
}

__global__ void k_post(const float* __restrict__ xcur, const float* __restrict__ sum,
                       const float* __restrict__ sumsq, const unsigned* __restrict__ mnk,
                       const unsigned* __restrict__ mxk, const float* __restrict__ deg,
                       const float* __restrict__ scal, const unsigned short* __restrict__ Bop,
                       const float* __restrict__ bpost, unsigned short* __restrict__ out,
                       int n, int l) {
    int tile = wave_tile();                 // SGPR
    int lane = threadIdx.x & 31;
    int base = tile * 16;                   // SGPR
    if (base >= n) return;
    int mrow = lane & 15;
    int half = lane >> 4;
    int kb = half * 8;
    bool full = (base + 16 <= n);           // scalar
    int nodeA = min(base + mrow, n - 1);
    float degv = deg[nodeA];
    float degc = fmaxf(degv, 1.f);
    float inv = 1.f / degc;
    float dlog = logf(degc + 1.f);
    float avg = scal[0];
    float s1 = dlog / avg, s2 = avg / dlog;
    for (int t = 0; t < TT; t++) {
        v8f acc0 = {}, acc1 = {};
        unsigned arow = (unsigned)nodeA * CC + (unsigned)(t * 32) + (unsigned)kb;
#pragma unroll
        for (int ks = 0; ks < 13; ks++) {   // unrolled: section switch resolves statically
            float fv[16];
            feat8(ks, arow,      xcur, sum, sumsq, mnk, mxk, degv, inv, s1, s2, fv);
            feat8(ks, arow + 16, xcur, sum, sumsq, mnk, mxk, degv, inv, s1, s2, fv + 8);
            v16bf A;
#pragma unroll
            for (int j = 0; j < 16; j++) A[j] = f2bf(fv[j]);
            unsigned bo = (unsigned)((((l * TT + t) * 13 + ks) * 2 + 0) * 512 + lane * 16);
            v16bf B0 = cat16(ldv8(Bop, bo), ldv8(Bop, bo + 8));
            v16bf B1 = cat16(ldv8(Bop, bo + 512), ldv8(Bop, bo + 520));
            acc0 = wmma_bf16(A, B0, acc0);
            acc1 = wmma_bf16(A, B1, acc1);
        }
#pragma unroll
        for (int nt = 0; nt < 2; nt++) {
            float bias = bpost[(l * TT + t) * 32 + nt * 16 + mrow];
            unsigned col = (unsigned)(t * 32 + nt * 16 + mrow);
            if (full) {
#pragma unroll
                for (int r = 0; r < 8; r++) {
                    unsigned nm = (unsigned)(base + half * 8 + r);
                    float v = (nt == 0 ? acc0[r] : acc1[r]) + bias;
                    out[nm * CC + col] = f2bfbits(v);
                }
            } else {
#pragma unroll
                for (int r = 0; r < 8; r++) {
                    int nm = base + half * 8 + r;
                    if (nm < n) {
                        float v = (nt == 0 ? acc0[r] : acc1[r]) + bias;
                        out[(unsigned)nm * CC + col] = f2bfbits(v);
                    }
                }
            }
        }
    }
}

// ---------------- tower-mix linear (128x128) ----------------
__global__ void k_lin(const unsigned short* __restrict__ pin, const unsigned short* __restrict__ Bop,
                      const float* __restrict__ blin, float* __restrict__ out, int n, int l) {
    int tile = wave_tile();                 // SGPR
    int lane = threadIdx.x & 31;
    int base = tile * 16;                   // SGPR
    if (base >= n) return;
    int mrow = lane & 15;
    int half = lane >> 4;
    int kb = half * 8;
    bool full = (base + 16 <= n);           // scalar
    int nodeA = min(base + mrow, n - 1);
    v16bf A[4];
#pragma unroll
    for (int ks = 0; ks < 4; ks++) {
        unsigned ao = (unsigned)nodeA * CC + (unsigned)(ks * 32 + kb);
        A[ks] = cat16(ldv8(pin, ao), ldv8(pin, ao + 16));
    }
#pragma unroll
    for (int nt = 0; nt < 8; nt++) {
        v8f c = {};
#pragma unroll
        for (int ks = 0; ks < 4; ks++) {
            unsigned bo = (unsigned)((((l * 4 + ks) * 8 + nt) * 512) + lane * 16);
            v16bf B = cat16(ldv8(Bop, bo), ldv8(Bop, bo + 8));
            c = wmma_bf16(A[ks], B, c);
        }
        float bias = blin[l * CC + nt * 16 + mrow];
        unsigned col = (unsigned)(nt * 16 + mrow);
        if (full) {
#pragma unroll
            for (int r = 0; r < 8; r++) {
                unsigned nm = (unsigned)(base + half * 8 + r);
                out[nm * CC + col] = c[r] + bias;
            }
        } else {
#pragma unroll
            for (int r = 0; r < 8; r++) {
                int nm = base + half * 8 + r;
                if (nm < n) out[(unsigned)nm * CC + col] = c[r] + bias;
            }
        }
    }
}

// ---------------- batch norm + relu ----------------
__global__ void k_bnzero(float* bn) {
    int i = threadIdx.x;
    if (i < 256) bn[i] = 0.f;
}
__global__ void k_bnstats(const float* __restrict__ x, float* bn, int n) {
    int c = threadIdx.x; // 128 threads
    float s = 0.f, s2 = 0.f;
    for (int i = blockIdx.x; i < n; i += gridDim.x) {
        float v = x[(unsigned)i * CC + c];
        s += v; s2 += v * v;
    }
    atomicAdd(&bn[c], s);
    atomicAdd(&bn[128 + c], s2);
}
__global__ void k_bnapply(float* x, const float* bn, const float* gamma, const float* beta,
                          int total, int l) {
    int i = blockIdx.x * blockDim.x + threadIdx.x;
    if (i >= total) return;
    int c = i & (CC - 1);
    float mu = bn[c] * (1.0f / NN);
    float var = bn[128 + c] * (1.0f / NN) - mu * mu;
    float v = (x[i] - mu) * rsqrtf(var + 1e-5f) * gamma[l * CC + c] + beta[l * CC + c];
    x[i] = fmaxf(v, 0.f);
}

// ---------------- MLP head + log_softmax ----------------
__global__ void k_head(const float* __restrict__ x, const float* W1, const float* B1,
                       const float* W2, const float* B2, const float* W3, const float* B3,
                       float* out) {
    __shared__ float xs[128], h1[96], h2[48], lg[16], red[2];
    int node = blockIdx.x, tid = threadIdx.x;
    xs[tid] = x[(unsigned)node * CC + tid];
    __syncthreads();
    if (tid < HH1) {
        float a = B1[tid];
        for (int k = 0; k < 128; k++) a += xs[k] * W1[k * HH1 + tid];
        h1[tid] = fmaxf(a, 0.f);
    }
    __syncthreads();
    if (tid < HH2) {
        float a = B2[tid];
        for (int k = 0; k < HH1; k++) a += h1[k] * W2[k * HH2 + tid];
        h2[tid] = fmaxf(a, 0.f);
    }
    __syncthreads();
    if (tid < NCLS) {
        float a = B3[tid];
        for (int k = 0; k < HH2; k++) a += h2[k] * W3[k * NCLS + tid];
        lg[tid] = a;
    }
    __syncthreads();
    if (tid == 0) {
        float m = lg[0];
        for (int j = 1; j < NCLS; j++) m = fmaxf(m, lg[j]);
        float s = 0.f;
        for (int j = 0; j < NCLS; j++) s += expf(lg[j] - m);
        red[0] = m; red[1] = logf(s);
    }
    __syncthreads();
    if (tid < NCLS) out[(unsigned)node * NCLS + tid] = lg[tid] - red[0] - red[1];
}

extern "C" void kernel_launch(void* const* d_in, const int* in_sizes, int n_in,
                              void* d_out, int out_size, void* d_ws, size_t ws_size,
                              hipStream_t stream) {
    const float* x0   = (const float*)d_in[0];
    const int*   ei   = (const int*)d_in[1];
    const int*   src  = ei;
    const int*   dst  = ei + EE;
    const float* Wpre = (const float*)d_in[2];
    const float* bpre = (const float*)d_in[3];
    const float* Wpost= (const float*)d_in[4];
    const float* bpost= (const float*)d_in[5];
    const float* Wlin = (const float*)d_in[6];
    const float* blin = (const float*)d_in[7];
    const float* gamma= (const float*)d_in[8];
    const float* beta = (const float*)d_in[9];
    const float* Wh1  = (const float*)d_in[10];
    const float* bh1  = (const float*)d_in[11];
    const float* Wh2  = (const float*)d_in[12];
    const float* bh2  = (const float*)d_in[13];
    const float* Wh3  = (const float*)d_in[14];
    const float* bh3  = (const float*)d_in[15];
    float* out = (float*)d_out;

    // carve workspace
    char* p = (char*)d_ws;
    auto carve = [&](size_t bytes) -> void* {
        void* r = (void*)p;
        p += (bytes + 255) & ~(size_t)255;
        return r;
    };
    float*          deg    = (float*)carve(NN * 4);
    float*          scal   = (float*)carve(64 * 4);
    float*          bn     = (float*)carve(256 * 4);
    unsigned short* BopPre = (unsigned short*)carve(DEPTH * TT * 2 * 2 * 512 * 2);
    unsigned short* BopPost= (unsigned short*)carve(DEPTH * TT * 13 * 2 * 512 * 2);
    unsigned short* BopLin = (unsigned short*)carve(DEPTH * 4 * 8 * 512 * 2);
    unsigned short* xb     = (unsigned short*)carve((size_t)NN * CC * 2);
    float*          bufA   = (float*)carve((size_t)NN * CC * 4);
    float*          bufB   = (float*)carve((size_t)NN * CC * 4);
    float*          sum    = (float*)carve((size_t)NN * CC * 4);
    float*          sumsq  = (float*)carve((size_t)NN * CC * 4);
    unsigned*       mnk    = (unsigned*)carve((size_t)NN * CC * 4);
    unsigned*       mxk    = (unsigned*)carve((size_t)NN * CC * 4);
    unsigned short* postout= (unsigned short*)carve((size_t)NN * CC * 2);

    const int total = NN * CC;

    k_init<<<(NN + 255) / 256, 256, 0, stream>>>(deg, scal, NN);
    k_deg<<<(EE + 255) / 256, 256, 0, stream>>>(dst, deg, EE);
    k_avglog<<<(NN + 255) / 256, 256, 0, stream>>>(deg, scal, NN);
    k_prep_pre<<<(DEPTH * TT * 2 * 2 * 512 + 255) / 256, 256, 0, stream>>>(Wpre, BopPre);
    k_prep_post<<<(DEPTH * TT * 13 * 2 * 512 + 255) / 256, 256, 0, stream>>>(Wpost, BopPost);
    k_prep_lin<<<(DEPTH * 4 * 8 * 512 + 255) / 256, 256, 0, stream>>>(Wlin, BopLin);

    int etiles  = (EE + 15) / 16;
    int eblocks = (etiles + 7) / 8;
    int ntiles  = (NN + 15) / 16;
    int nblocks = (ntiles + 7) / 8;

    for (int l = 0; l < DEPTH; l++) {
        const float* xcur = (l == 0) ? x0 : ((l & 1) ? bufA : bufB);
        float*       xnew = (l & 1) ? bufB : bufA;

        k_cvt_init<<<(total + 255) / 256, 256, 0, stream>>>(xcur, xb, sum, sumsq, mnk, mxk, total);
        k_msg<<<eblocks, 256, 0, stream>>>(xb, BopPre, bpre, src, dst, sum, sumsq, mnk, mxk, EE, l);
        k_post<<<nblocks, 256, 0, stream>>>(xcur, sum, sumsq, mnk, mxk, deg, scal,
                                            BopPost, bpost, postout, NN, l);
        k_lin<<<nblocks, 256, 0, stream>>>(postout, BopLin, blin, xnew, NN, l);
        k_bnzero<<<1, 256, 0, stream>>>(bn);
        k_bnstats<<<256, 128, 0, stream>>>(xnew, bn, NN);
        k_bnapply<<<(total + 255) / 256, 256, 0, stream>>>(xnew, bn, gamma, beta, total, l);
    }
    // DEPTH=4 -> final activations in bufB
    k_head<<<NN, 128, 0, stream>>>(bufB, Wh1, bh1, Wh2, bh2, Wh3, bh3, out);
}
